// GINModel_26044681683192
// MI455X (gfx1250) — matrix operand order repaired and checked by reading, not compiled, not run on hardware
//
#include <hip/hip_runtime.h>
#include <hip/hip_bf16.h>
#include <math.h>

// GIN model for MI455X (gfx1250, wave32).
//   - GEMMs ([N,64]x[64,64]) via v_wmma_f32_16x16x32_f16, f32 accumulate.
//   - Edge aggregation via feature-parallel global f32 atomics (L2-resident:
//     h is 25.6MB << 192MB L2, so scatter is L2-bound, not HBM-bound).
//   - BatchNorm: two-pass (atomic column sum/sumsq, then fused norm+act).
// Workspace: 3 * N*64 f32 buffers + f16 weights + stats/pool ~= 77 MB.

typedef __attribute__((ext_vector_type(16))) _Float16 v16h;
typedef __attribute__((ext_vector_type(8)))  float    v8f;

#define HID      64
#define NLAYERS  4
#define GRAPHS   512
#define BN_EPS   1e-5f

__device__ __forceinline__ float gelu_exact(float x) {
  return 0.5f * x * (1.0f + erff(x * 0.70710678118654752f));
}

// -------------------------------------------------------------------------
__global__ void zero_f32(float* __restrict__ p, int n) {
  int i = blockIdx.x * blockDim.x + threadIdx.x;
  if (i < n) p[i] = 0.0f;
}

// pad x [N,6] -> h [N,64] (cols 6..63 = 0 so layer0 shares the K=64 path)
__global__ void load_x(const float* __restrict__ x, float* __restrict__ h, int N) {
  int i = blockIdx.x * blockDim.x + threadIdx.x;
  if (i >= N * HID) return;
  int r = i >> 6, f = i & 63;
  h[i] = (f < 6) ? x[r * 6 + f] : 0.0f;
}

// z = (1 + eps) * h   (eps is a trained device scalar)
__global__ void zinit(const float* __restrict__ h, float* __restrict__ z,
                      const float* __restrict__ epsp, int n) {
  int i = blockIdx.x * blockDim.x + threadIdx.x;
  if (i >= n) return;
  z[i] = (1.0f + epsp[0]) * h[i];
}

// z[dst,f] += h[src,f] for every edge; one atomic per (edge, feature)
__global__ void scatter_add(const float* __restrict__ h, const int* __restrict__ ei,
                            int E, int D, float* __restrict__ z) {
  int t = blockIdx.x * blockDim.x + threadIdx.x;
  if (t >= E * D) return;
  int e, f;
  if (D == HID) { e = t >> 6; f = t & 63; }
  else          { e = t / D;  f = t - e * D; }
  int s = ei[e];
  int d = ei[E + e];
  atomicAdd(&z[(size_t)d * HID + f], h[(size_t)s * HID + f]);
}

// float weights [dk,64] -> f16 [64,64], zero-padded rows dk..63
__global__ void convert_w(const float* __restrict__ W, int dk,
                          _Float16* __restrict__ out) {
  int i = blockIdx.x * blockDim.x + threadIdx.x;
  if (i >= HID * HID) return;
  int k = i >> 6, n = i & 63;
  out[i] = (k < dk) ? (_Float16)W[(size_t)k * HID + n] : (_Float16)0.0f;
}

// -------------------------------------------------------------------------
// OUT[N,64] = f16(Z[N,64]) @ Wf16[64,64] + bias, WMMA 16x16x32 f16, f32 accum.
// Per wave: one 16-row tile x all 64 cols. B fragments preloaded once.
// A layout  (contiguous, matches B): lane L -> row M=L%16, koff=(L/16)*16,
//   halves i: K = koff + 2i.
// B layout  (SWMMAC B pattern): lane L -> col N=L%16, kbase=(L/16)*16,
//   VGPR j holds rows kbase+2j, kbase+2j+1.
// C layout: VGPR v -> rows v (lanes 0-15) and v+8 (lanes 16-31).
// Stores: full-tile fast path is branch-free (N%16==0 here); rare tail path
// keeps per-row predication for generality.
__global__ void __launch_bounds__(128)
gemm64_wmma(const float* __restrict__ Z, const _Float16* __restrict__ W,
            const float* __restrict__ bias, float* __restrict__ OUT, int N)
{
  const int lane = threadIdx.x & 31;
  const int wave = threadIdx.x >> 5;
  const int nloc = lane & 15;
  const int hi   = lane >> 4;

  v16h B[2][4];
#pragma unroll
  for (int kb = 0; kb < 2; ++kb) {
    const int kbase = kb * 32 + hi * 16;
#pragma unroll
    for (int nb = 0; nb < 4; ++nb) {
      const _Float16* wp = W + (size_t)kbase * HID + nb * 16 + nloc;
      v16h b;
#pragma unroll
      for (int j = 0; j < 8; ++j) {
        b[2 * j]     = wp[(2 * j) * HID];
        b[2 * j + 1] = wp[(2 * j + 1) * HID];
      }
      B[kb][nb] = b;
    }
  }

  float bv[4];
#pragma unroll
  for (int nb = 0; nb < 4; ++nb) bv[nb] = bias[nb * 16 + nloc];

  const v8f zc = {0.f, 0.f, 0.f, 0.f, 0.f, 0.f, 0.f, 0.f};
  const int ntiles = (N + 15) >> 4;

  for (int t = blockIdx.x * 4 + wave; t < ntiles; t += (int)(gridDim.x * 4)) {
    const int row0 = t << 4;
    int ra = row0 + (lane & 15);
    if (ra > N - 1) ra = N - 1;            // clamp (keeps EXEC all-1 for WMMA)
    const float* zr = Z + (size_t)ra * HID;

    v16h A0, A1;
#pragma unroll
    for (int i = 0; i < 8; ++i) {
      const int kk = hi * 16 + 2 * i;
      A0[2 * i]     = (_Float16)zr[kk];
      A0[2 * i + 1] = (_Float16)zr[kk + 1];
      A1[2 * i]     = (_Float16)zr[32 + kk];
      A1[2 * i + 1] = (_Float16)zr[32 + kk + 1];
    }

    v8f C[4];
#pragma unroll
    for (int nb = 0; nb < 4; ++nb) C[nb] = zc;

#pragma unroll
    for (int nb = 0; nb < 4; ++nb) {
      C[nb] = __builtin_amdgcn_wmma_f32_16x16x32_f16(
                  false, A0, false, B[0][nb], (short)0, C[nb], false, false);
      C[nb] = __builtin_amdgcn_wmma_f32_16x16x32_f16(
                  false, A1, false, B[1][nb], (short)0, C[nb], false, false);
    }

    if (row0 + 16 <= N) {
      // full tile: unconditional clause-friendly stores, one base address
      float* op = OUT + (size_t)(row0 + hi * 8) * HID + nloc;
#pragma unroll
      for (int v = 0; v < 8; ++v) {
        float* o = op + (size_t)v * HID;
        o[0]  = C[0][v] + bv[0];
        o[16] = C[1][v] + bv[1];
        o[32] = C[2][v] + bv[2];
        o[48] = C[3][v] + bv[3];
      }
    } else {
      // ragged tail tile
#pragma unroll
      for (int v = 0; v < 8; ++v) {
        const int rr = row0 + v + hi * 8;
        if (rr < N) {
          float* o = OUT + (size_t)rr * HID + nloc;
          o[0]  = C[0][v] + bv[0];
          o[16] = C[1][v] + bv[1];
          o[32] = C[2][v] + bv[2];
          o[48] = C[3][v] + bv[3];
        }
      }
    }
  }
}

// -------------------------------------------------------------------------
// per-column sum / sumsq over N rows -> stats[0..63]=sum, stats[64..127]=sumsq
__global__ void bn_stats(const float* __restrict__ X, int N,
                         float* __restrict__ stats) {
  const int c = threadIdx.x & 63;
  float s = 0.0f, ss = 0.0f;
  for (int r = blockIdx.x * 4 + (threadIdx.x >> 6); r < N; r += (int)(gridDim.x * 4)) {
    float v = X[(size_t)r * HID + c];
    s += v;
    ss += v * v;
  }
  atomicAdd(&stats[c], s);
  atomicAdd(&stats[64 + c], ss);
}

// fused BN + activation; relu_pool==1 additionally accumulates the graph pool
__global__ void bn_act(const float* __restrict__ in, float* __restrict__ out,
                       const float* __restrict__ gamma, const float* __restrict__ beta,
                       const float* __restrict__ stats, int N, float invN,
                       int relu_pool, const int* __restrict__ batch,
                       float* __restrict__ pooled, int loff) {
  int i = blockIdx.x * blockDim.x + threadIdx.x;
  if (i >= N * HID) return;
  const int c = i & 63;
  const int r = i >> 6;
  const float mean = stats[c] * invN;
  const float var  = fmaxf(stats[64 + c] * invN - mean * mean, 0.0f);
  const float sc   = gamma[c] * rsqrtf(var + BN_EPS);
  float v = (in[i] - mean) * sc + beta[c];
  if (relu_pool) {
    v = fmaxf(v, 0.0f);
    out[i] = v;
    atomicAdd(&pooled[(size_t)batch[r] * (HID * NLAYERS) + loff + c], v);
  } else {
    out[i] = gelu_exact(v);
  }
}

__global__ void count_nodes(const int* __restrict__ batch, int N,
                            float* __restrict__ counts) {
  int i = blockIdx.x * blockDim.x + threadIdx.x;
  if (i < N) atomicAdd(&counts[batch[i]], 1.0f);
}

// head: y[g] = gelu(emb @ W1 + b1) @ W2 + b2, one 64-thread block per graph
__global__ void __launch_bounds__(64)
head_kernel(const float* __restrict__ pooled, const float* __restrict__ counts,
            const float* __restrict__ W1, const float* __restrict__ b1,
            const float* __restrict__ W2, const float* __restrict__ b2,
            float* __restrict__ y) {
  __shared__ float emb[HID * NLAYERS];
  __shared__ float hid[HID];
  const int g = blockIdx.x;
  const float inv = 1.0f / fmaxf(counts[g], 1.0f);
  for (int k = threadIdx.x; k < HID * NLAYERS; k += 64)
    emb[k] = pooled[(size_t)g * (HID * NLAYERS) + k] * inv;
  __syncthreads();
  const int j = threadIdx.x;
  float acc = b1[j];
  for (int k = 0; k < HID * NLAYERS; ++k) acc += emb[k] * W1[k * HID + j];
  hid[j] = gelu_exact(acc) * W2[j];
  __syncthreads();
  if (j == 0) {
    float s = b2[0];
    for (int k = 0; k < HID; ++k) s += hid[k];
    y[g] = s;
  }
}

// -------------------------------------------------------------------------
extern "C" void kernel_launch(void* const* d_in, const int* in_sizes, int n_in,
                              void* d_out, int out_size, void* d_ws, size_t ws_size,
                              hipStream_t stream) {
  const float* x     = (const float*)d_in[0];
  const int*   ei    = (const int*)d_in[1];
  const int*   batch = (const int*)d_in[2];
  const int N = in_sizes[0] / 6;
  const int E = in_sizes[1] / 2;

  // params flattened pytree-style (sorted dict keys): head first, then layers
  const float* hW1 = (const float*)d_in[4];  // [256,64]
  const float* hW2 = (const float*)d_in[5];  // [64,1]
  const float* hb1 = (const float*)d_in[6];  // [64]
  const float* hb2 = (const float*)d_in[7];  // [1]
  // layer i leaves at 8+9*i: W1,W2,b1,b2,be1,be2,eps,g1,g2

  // workspace carve
  float*     h      = (float*)d_ws;
  float*     zb     = h  + (size_t)N * HID;
  float*     tb     = zb + (size_t)N * HID;
  _Float16*  wf     = (_Float16*)(tb + (size_t)N * HID);
  float*     stats  = (float*)(wf + (size_t)2 * NLAYERS * HID * HID);
  float*     pooled = stats + 128;
  float*     counts = pooled + (size_t)GRAPHS * HID * NLAYERS;

  const int TB = 256;
  const int nelems = N * HID;
  const int gelems = (nelems + TB - 1) / TB;

  for (int i = 0; i < NLAYERS; ++i) {
    const float* W1 = (const float*)d_in[8 + 9 * i + 0];
    const float* W2 = (const float*)d_in[8 + 9 * i + 1];
    convert_w<<<(HID * HID + TB - 1) / TB, TB, 0, stream>>>(
        W1, (i == 0) ? 6 : HID, wf + (size_t)(2 * i) * HID * HID);
    convert_w<<<(HID * HID + TB - 1) / TB, TB, 0, stream>>>(
        W2, HID, wf + (size_t)(2 * i + 1) * HID * HID);
  }

  load_x<<<gelems, TB, 0, stream>>>(x, h, N);
  zero_f32<<<(GRAPHS * (HID * NLAYERS + 1) + TB - 1) / TB, TB, 0, stream>>>(
      pooled, GRAPHS * (HID * NLAYERS + 1));   // pooled + counts contiguous
  count_nodes<<<(N + TB - 1) / TB, TB, 0, stream>>>(batch, N, counts);

  const int ntiles     = (N + 15) / 16;
  const int gemmBlocks = (ntiles + 3) / 4;
  const float invN = 1.0f / (float)N;

  for (int i = 0; i < NLAYERS; ++i) {
    const float* b1  = (const float*)d_in[8 + 9 * i + 2];
    const float* b2  = (const float*)d_in[8 + 9 * i + 3];
    const float* be1 = (const float*)d_in[8 + 9 * i + 4];
    const float* be2 = (const float*)d_in[8 + 9 * i + 5];
    const float* eps = (const float*)d_in[8 + 9 * i + 6];
    const float* g1  = (const float*)d_in[8 + 9 * i + 7];
    const float* g2  = (const float*)d_in[8 + 9 * i + 8];

    // z = (1+eps)*h + scatter_add(h[src] -> dst)
    zinit<<<gelems, TB, 0, stream>>>(h, zb, eps, nelems);
    const int D = (i == 0) ? 6 : HID;   // layer0 cols 6..63 are zero anyway
    scatter_add<<<(E * D + TB - 1) / TB, TB, 0, stream>>>(h, ei, E, D, zb);

    // t = z @ W1 + b1 ; BN ; GELU (in-place)
    gemm64_wmma<<<gemmBlocks, 128, 0, stream>>>(
        zb, wf + (size_t)(2 * i) * HID * HID, b1, tb, N);
    zero_f32<<<1, 128, 0, stream>>>(stats, 128);
    bn_stats<<<512, 256, 0, stream>>>(tb, N, stats);
    bn_act<<<gelems, TB, 0, stream>>>(tb, tb, g1, be1, stats, N, invN,
                                      0, nullptr, nullptr, 0);

    // z = t @ W2 + b2 ; BN ; ReLU -> h (+ pooled accumulation)
    gemm64_wmma<<<gemmBlocks, 128, 0, stream>>>(
        tb, wf + (size_t)(2 * i + 1) * HID * HID, b2, zb, N);
    zero_f32<<<1, 128, 0, stream>>>(stats, 128);
    bn_stats<<<512, 256, 0, stream>>>(zb, N, stats);
    bn_act<<<gelems, TB, 0, stream>>>(zb, h, g2, be2, stats, N, invN,
                                      1, batch, pooled, i * HID);
  }

  head_kernel<<<GRAPHS, 64, 0, stream>>>(pooled, counts, hW1, hb1, hW2, hb2,
                                         (float*)d_out);
}
